// MessageFunction_31327491457165
// MI455X (gfx1250) — matrix operand order repaired and checked by reading, not compiled, not run on hardware
//
#include <hip/hip_runtime.h>
#include <hip/hip_bf16.h>

// Problem constants (match reference)
#define BB 32
#define NN 512
#define DD 256
#define EE 8
#define KK (EE * DD)        // 2048 reduction dim of the fused GEMM
#define ROWS (BB * NN)      // 16384 output rows
#define KBLK (KK / 32)      // 64 WMMA K-steps
#define STRIP_BYTES (16 * KK * 2)   // 64KiB: one 16-row bf16 A strip

typedef __attribute__((ext_vector_type(8)))  __bf16 v8bf;
typedef __attribute__((ext_vector_type(16))) __bf16 v16bf;
typedef __attribute__((ext_vector_type(8)))  float  v8f;

__device__ __forceinline__ unsigned short f2bf_rne(float f) {
    unsigned int u = __float_as_uint(f);
    unsigned int r = u + 0x7FFFu + ((u >> 16) & 1u);   // round-to-nearest-even
    return (unsigned short)(r >> 16);
}

// CDNA5 async copy: LDS[lds_off] = MEM[gaddr], 16 bytes, tracked by ASYNCcnt.
__device__ __forceinline__ void async_b128_to_lds(unsigned lds_off,
                                                  const void* gaddr) {
    asm volatile("global_load_async_to_lds_b128 %0, %1, off"
                 :: "v"(lds_off), "v"((unsigned long long)(uintptr_t)gaddr)
                 : "memory");
}
__device__ __forceinline__ void wait_asynccnt0() {
    asm volatile("s_wait_asynccnt 0x0" ::: "memory");
}

// ---------------------------------------------------------------------------
// Kernel 1: pack matrix_in/matrix_out (f32 [E,D,D]) into bf16 B-fragment
// layout: Bpack[((side*16 + t)*64 + kb)*32 + lane][q] with
//   n = lane%16 (output col in tile t), krel = (lane/16)*16 + q, k = kb*32+krel
//   element = W[e, i, j] with e = k/256, j = k%256, i = t*16 + n
// ---------------------------------------------------------------------------
__global__ void pack_w_kernel(const float* __restrict__ w_in,
                              const float* __restrict__ w_out,
                              unsigned short* __restrict__ bpack) {
    unsigned id = blockIdx.x * blockDim.x + threadIdx.x;   // [0, 2*16*64*32*16)
    const unsigned per_side = 16u * 64u * 32u * 16u;       // 524288
    unsigned side = id / per_side;
    unsigned rem  = id % per_side;
    unsigned t    = rem / (64u * 32u * 16u);
    unsigned rem2 = rem % (64u * 32u * 16u);
    unsigned kb   = rem2 / (32u * 16u);
    unsigned rem3 = rem2 % (32u * 16u);
    unsigned lane = rem3 / 16u;
    unsigned q    = rem3 % 16u;

    unsigned n    = lane & 15u;
    unsigned krel = (lane >> 4) * 16u + q;
    unsigned k    = kb * 32u + krel;
    unsigned e    = k >> 8;
    unsigned j    = k & 255u;
    unsigned i    = t * 16u + n;

    const float* w = side ? w_out : w_in;
    float v = w[(e * DD + i) * DD + j];
    bpack[id] = f2bf_rne(v);
}

// ---------------------------------------------------------------------------
// Kernel 2: typed bucket aggregation.  Block = one (b, v); thread tid owns
// feature column j = tid (exclusive LDS addresses -> no atomics).
// S[(b*N + v)*2048 + e*256 + j] in bf16.
// ---------------------------------------------------------------------------
__global__ void aggregate_kernel(const float* __restrict__ node_state,
                                 const int* __restrict__ adj,
                                 unsigned short* __restrict__ s_in,
                                 unsigned short* __restrict__ s_out) {
    __shared__ int   cls_row[NN];
    __shared__ int   cls_col[NN];
    __shared__ float acc_in [EE * DD];
    __shared__ float acc_out[EE * DD];

    const int v   = blockIdx.x;
    const int b   = blockIdx.y;
    const int tid = threadIdx.x;           // 256 threads

    const size_t adj_b = (size_t)b * NN * NN;
    cls_row[tid]        = adj[adj_b + (size_t)v * NN + tid];
    cls_row[tid + 256]  = adj[adj_b + (size_t)v * NN + tid + 256];
    cls_col[tid]        = adj[adj_b + (size_t)tid * NN + v];
    cls_col[tid + 256]  = adj[adj_b + (size_t)(tid + 256) * NN + v];
#pragma unroll
    for (int r = 0; r < EE; ++r) {
        acc_in [r * DD + tid] = 0.0f;
        acc_out[r * DD + tid] = 0.0f;
    }
    __syncthreads();

    const float* ns = node_state + (size_t)b * NN * DD + tid;
    for (int w = 0; w < NN; ++w) {
        float x  = ns[(size_t)w * DD];
        int   ci = cls_row[w];
        int   co = cls_col[w];
        acc_in [ci * DD + tid] += x;
        acc_out[co * DD + tid] += x;
    }
    __syncthreads();

    const size_t base = (size_t)(b * NN + v) * KK;
#pragma unroll
    for (int r = 0; r < EE; ++r) {
        s_in [base + r * DD + tid] = f2bf_rne(acc_in [r * DD + tid]);
        s_out[base + r * DD + tid] = f2bf_rne(acc_out[r * DD + tid]);
    }
}

// ---------------------------------------------------------------------------
// Kernel 3: WMMA GEMM.
//  * Both 16-row A strips (S_in/S_out, 64KiB each, contiguous) staged in LDS
//    once per block with global_load_async_to_lds_b128 (ASYNCcnt), shared by
//    the 4 waves of each side; A fragments then come from ds_load_b128.
//  * B fragments register double-buffered; sched_barrier(0) fences keep the
//    next K-step's loads in flight across the current WMMA group so waits
//    become partial instead of s_wait_loadcnt 0 per WMMA.
// ---------------------------------------------------------------------------
__global__ void __launch_bounds__(256)
wmma_gemm_kernel(const unsigned short* __restrict__ s_in,
                 const unsigned short* __restrict__ s_out,
                 const unsigned short* __restrict__ bpack,
                 const float* __restrict__ bias,
                 float* __restrict__ out) {
    __shared__ __bf16 sA[2 * 16 * KK];     // 128 KiB: [side][row 16][k 2048]

    const int tid    = threadIdx.x;
    const int lane   = tid & 31;
    const int waveId = tid >> 5;           // 0..7
    const int side   = waveId >> 2;        // 0 = in, 1 = out
    const int g      = waveId & 3;         // 64-col group within the side

    const int m16 = lane & 15;             // A row within tile / N col index
    const int kh  = lane >> 4;             // A K-half select

    // ---- Stage A strips into LDS (async, cooperative, fully coalesced) ----
    {
        const char* gin  = (const char*)(s_in  + (size_t)blockIdx.x * 16 * KK);
        const char* gout = (const char*)(s_out + (size_t)blockIdx.x * 16 * KK);
        unsigned lds0 = (unsigned)(uintptr_t)(&sA[0]);          // LDS offset
        unsigned tb   = tid * 16;
#pragma unroll 4
        for (int i = 0; i < STRIP_BYTES / (256 * 16); ++i) {    // 16 iters
            unsigned off = i * 4096 + tb;
            async_b128_to_lds(lds0 + off,               gin  + off);
            async_b128_to_lds(lds0 + STRIP_BYTES + off, gout + off);
        }
        wait_asynccnt0();
        __syncthreads();
    }

    const __bf16* arow = sA + side * 16 * KK + m16 * KK;
    const __bf16* bbase = (const __bf16*)(bpack) +
        ((size_t)(side * 16 + g * 4) * 64 * 512 + lane * 16);

    auto loadA = [&](int kb) -> v16bf {                 // two ds_load_b128
        const __bf16* rp = arow + kb * 32 + kh * 16;
        v8bf lo = *(const v8bf*)(rp);
        v8bf hi = *(const v8bf*)(rp + 8);
        return __builtin_shufflevector(lo, hi,
                   0, 1, 2, 3, 4, 5, 6, 7, 8, 9, 10, 11, 12, 13, 14, 15);
    };
    auto loadB = [&](int kb, int t4) -> v16bf {         // one 32B global load
        return *(const v16bf*)(bbase + ((size_t)t4 * 64 + kb) * 512);
    };
#define WMMA_BF16(A, Bf, C) \
    __builtin_amdgcn_wmma_f32_16x16x32_bf16(false, (A), false, (Bf), (short)0, \
                                            (C), false, false)

    v8f acc0 = {}, acc1 = {}, acc2 = {}, acc3 = {};

    v16bf p0 = loadB(0, 0), p1 = loadB(0, 1), p2 = loadB(0, 2), p3 = loadB(0, 3);
    v16bf q0, q1, q2, q3;

    for (int kb = 0; kb < KBLK; kb += 2) {
        // Stage kb+1's B fragments while kb's feed the matrix pipe.
        q0 = loadB(kb + 1, 0); q1 = loadB(kb + 1, 1);
        q2 = loadB(kb + 1, 2); q3 = loadB(kb + 1, 3);
        __builtin_amdgcn_sched_barrier(0);
        {
            v16bf a = loadA(kb);
            acc0 = WMMA_BF16(a, p0, acc0);
            acc1 = WMMA_BF16(a, p1, acc1);
            acc2 = WMMA_BF16(a, p2, acc2);
            acc3 = WMMA_BF16(a, p3, acc3);
        }
        __builtin_amdgcn_sched_barrier(0);
        const int k2 = (kb + 2 < KBLK) ? kb + 2 : 0;   // tail: dummy reload
        p0 = loadB(k2, 0); p1 = loadB(k2, 1);
        p2 = loadB(k2, 2); p3 = loadB(k2, 3);
        __builtin_amdgcn_sched_barrier(0);
        {
            v16bf a = loadA(kb + 1);
            acc0 = WMMA_BF16(a, q0, acc0);
            acc1 = WMMA_BF16(a, q1, acc1);
            acc2 = WMMA_BF16(a, q2, acc2);
            acc3 = WMMA_BF16(a, q3, acc3);
        }
        __builtin_amdgcn_sched_barrier(0);
    }

    // Epilogue: bias + store. C/D layout: VGPR r, lane -> (m = r + 8*kh, n = m16)
    const size_t row0 = (size_t)blockIdx.x * 16;
#pragma unroll
    for (int t4 = 0; t4 < 4; ++t4) {
        const int col = side * DD + (g * 4 + t4) * 16 + m16;
        const float bv = bias[col];
        const v8f acc = t4 == 0 ? acc0 : t4 == 1 ? acc1 : t4 == 2 ? acc2 : acc3;
#pragma unroll
        for (int r = 0; r < 8; ++r) {
            int m = r + 8 * kh;
            out[(row0 + m) * (2 * DD) + col] = acc[r] + bv;
        }
    }
#undef WMMA_BF16
}

// ---------------------------------------------------------------------------
extern "C" void kernel_launch(void* const* d_in, const int* in_sizes, int n_in,
                              void* d_out, int out_size, void* d_ws, size_t ws_size,
                              hipStream_t stream) {
    const float* node_state = (const float*)d_in[0];   // [B,N,D] f32
    const int*   adj        = (const int*)  d_in[1];   // [B,N,N] i32
    const float* w_in       = (const float*)d_in[2];   // [E,D,D] f32
    const float* w_out      = (const float*)d_in[3];   // [E,D,D] f32
    const float* bias       = (const float*)d_in[4];   // [2D]    f32
    float*       out        = (float*)d_out;           // [B,N,2D] f32

    // Workspace: S_in (64MiB) | S_out (64MiB) | Bpack (2MiB)  -> 130MiB
    unsigned short* s_in  = (unsigned short*)d_ws;
    unsigned short* s_out = s_in  + (size_t)ROWS * KK;
    unsigned short* bpack = s_out + (size_t)ROWS * KK;

    pack_w_kernel<<<dim3(2 * 16 * 64 * 32 * 16 / 256), dim3(256), 0, stream>>>(
        w_in, w_out, bpack);

    aggregate_kernel<<<dim3(NN, BB), dim3(256), 0, stream>>>(
        node_state, adj, s_in, s_out);

    wmma_gemm_kernel<<<dim3(ROWS / 16), dim3(256), 0, stream>>>(
        s_in, s_out, bpack, bias, out);
}
